// GraphNet_87514253623335
// MI455X (gfx1250) — compile-verified, matrix operand order checked by reading, not run on hardware
//
#include <hip/hip_runtime.h>

// ---------------------------------------------------------------------------
// GNN (2x GINEConv-style layers) for MI455X / gfx1250.
// Edge message GEMMs + node MLPs run on v_wmma_f32_16x16x32_bf16 (wave32),
// aggregation via global_atomic_add_f32. bf16 operands, f32 accumulation.
// ---------------------------------------------------------------------------

typedef __attribute__((ext_vector_type(16))) __bf16 bf16x16;
typedef __attribute__((ext_vector_type(8)))  float  floatx8;

#define N_NODES 50000
#define N_EDGES 800000
#define FDIM    100
#define KP3F    320          // 3*F = 300 padded to multiple of 32
#define OUT_NT  7            // output tiles: 112 cols (100 padded)
#define NODE_KP 128          // F=100 padded to 128 for node MLP K

union FragU {
    uint4   q[2];
    bf16x16 v;
};

__device__ __forceinline__ unsigned short f2bf(float f) {
    unsigned int u = __float_as_uint(f);
    u += 0x7fffu + ((u >> 16) & 1u);            // round-to-nearest-even
    return (unsigned short)(u >> 16);
}

// A/B 16-bit WMMA fragment load (ISA 7.12.2 layout): lane l holds row/col
// (l&15); halves 0..7 = K = kb..kb+7, halves 8..15 = K = 16+kb..16+kb+7,
// kb = 8*(l>=16). Two 16B loads per fragment.
__device__ __forceinline__ bf16x16 load_frag(const unsigned short* row, int kbase, int lane) {
    int kb = ((lane >> 4) & 1) << 3;
    FragU u;
    u.q[0] = *reinterpret_cast<const uint4*>(row + kbase + kb);
    u.q[1] = *reinterpret_cast<const uint4*>(row + kbase + 16 + kb);
    return u.v;
}

__device__ __forceinline__ void lds_fence() {
    asm volatile("s_wait_dscnt 0x0" ::: "memory");
}

// --------------------------- weight conversion -----------------------------
// w[k][out] (f32) -> wT[o][k] (bf16), zero padded to [OUTP][KP].
__global__ void convert_wT_kernel(const float* __restrict__ w, unsigned short* __restrict__ wT,
                                  int K, int OUT, int KP, int OUTP) {
    int idx = blockIdx.x * blockDim.x + threadIdx.x;
    int total = KP * OUTP;
    if (idx >= total) return;
    int o = idx / KP, k = idx - o * KP;
    float v = (o < OUT && k < K) ? w[(long)k * OUT + o] : 0.f;
    wT[(long)o * KP + k] = f2bf(v);
}

__global__ void copy4_kernel(const float* __restrict__ src, float* __restrict__ dst, int n4) {
    int i = blockIdx.x * blockDim.x + threadIdx.x;
    if (i < n4) reinterpret_cast<float4*>(dst)[i] = reinterpret_cast<const float4*>(src)[i];
}

// ------------------------------ edge kernel --------------------------------
// One wave per 16-edge tile: m = relu([x_i|e|x_j] @ W + b) then atomic
// scatter-add into agg[dst]. agg pre-initialized to x (=> agg + x).
__global__ __launch_bounds__(128) void edge_msg_kernel(
    const float* __restrict__ x, const int* __restrict__ src_idx,
    const int* __restrict__ dst_idx, const float* __restrict__ eattr,
    const unsigned short* __restrict__ wT /*[112][320] bf16*/,
    const float* __restrict__ bias /*[100]*/, float* __restrict__ agg)
{
    __shared__ __align__(16) unsigned short Asm[4][16 * KP3F];   // 40 KB
    __shared__ int dsm[4][16];

    int wv = threadIdx.x >> 5, lane = threadIdx.x & 31;
    int tile = blockIdx.x * 4 + wv;
    if (tile >= N_EDGES / 16) return;
    unsigned short* A = Asm[wv];
    int* dl = dsm[wv];

    // ---- gather [16 x 320] bf16 A-tile: 2 lanes per edge row ----
    int row = lane >> 1, par = lane & 1;
    int e = tile * 16 + row;
    int s = src_idx[e], d = dst_idx[e];
    if (par == 0) dl[row] = d;
    const float* segs[3] = { x + (long)d * FDIM, eattr + (long)e * FDIM, x + (long)s * FDIM };
    unsigned short* Ar = A + row * KP3F;
#pragma unroll
    for (int sg = 0; sg < 3; ++sg) {
        const float* sp = segs[sg];
        unsigned short* dp = Ar + sg * FDIM;
        for (int j = par; j < 25; j += 2) {           // 25 float4 = 100 floats
            float4 v = reinterpret_cast<const float4*>(sp)[j];
            unsigned int lo = (unsigned)f2bf(v.x) | ((unsigned)f2bf(v.y) << 16);
            unsigned int hi = (unsigned)f2bf(v.z) | ((unsigned)f2bf(v.w) << 16);
            *reinterpret_cast<uint2*>(dp + j * 4) = make_uint2(lo, hi);
        }
    }
    for (int j = par; j < 5; j += 2)                  // zero pad cols 300..319
        *reinterpret_cast<uint2*>(Ar + 300 + j * 4) = make_uint2(0u, 0u);
    lds_fence();

    // ---- GEMM: [16x320] x [320x112] via 10 K-steps, 7 N-tiles ----
    int l15 = lane & 15;
    int mhi = ((lane >> 4) & 1) << 3;
    floatx8 acc[OUT_NT] = {};
    const unsigned short* Arow = A + l15 * KP3F;
#pragma unroll
    for (int k = 0; k < 10; ++k) {
        bf16x16 a = load_frag(Arow, k * 32, lane);
#pragma unroll
        for (int t = 0; t < OUT_NT; ++t) {
            bf16x16 b = load_frag(wT + (long)(t * 16 + l15) * KP3F, k * 32, lane);
            acc[t] = __builtin_amdgcn_wmma_f32_16x16x32_bf16(
                false, a, false, b, (short)0, acc[t], false, false);
        }
    }

    // ---- bias + relu + atomic scatter-add ----
#pragma unroll
    for (int t = 0; t < OUT_NT; ++t) {
        int col = t * 16 + l15;
        if (col < FDIM) {
            float bv = bias[col];
#pragma unroll
            for (int i = 0; i < 8; ++i) {
                int m = i + mhi;
                float v = acc[t][i] + bv;
                v = v > 0.f ? v : 0.f;
                atomicAdd(agg + (long)dl[m] * FDIM + col, v);
            }
        }
    }
}

// ------------------------------ node MLP kernel ----------------------------
// out = relu( relu(in @ w1 + b1) @ w2 + b2 ), one wave per 16-node tile.
__global__ __launch_bounds__(128) void node_mlp_kernel(
    const float* __restrict__ in, const unsigned short* __restrict__ w1T,
    const float* __restrict__ b1, const unsigned short* __restrict__ w2T,
    const float* __restrict__ b2, float* __restrict__ out,
    int H, int HPtiles, int K2P, int K2tiles)
{
    __shared__ __align__(16) unsigned short Asm[4][16 * NODE_KP];  // 16 KB
    __shared__ __align__(16) unsigned short Hsm[4][16 * NODE_KP];  // 16 KB

    int wv = threadIdx.x >> 5, lane = threadIdx.x & 31;
    int tile = blockIdx.x * 4 + wv;
    if (tile >= N_NODES / 16) return;
    unsigned short* A  = Asm[wv];
    unsigned short* Hb = Hsm[wv];

    // ---- gather [16 x 128] bf16 input tile ----
    int row = lane >> 1, par = lane & 1;
    const float* ir = in + (long)(tile * 16 + row) * FDIM;
    unsigned short* Ar = A + row * NODE_KP;
    for (int j = par; j < 25; j += 2) {
        float4 v = reinterpret_cast<const float4*>(ir)[j];
        unsigned int lo = (unsigned)f2bf(v.x) | ((unsigned)f2bf(v.y) << 16);
        unsigned int hi = (unsigned)f2bf(v.z) | ((unsigned)f2bf(v.w) << 16);
        *reinterpret_cast<uint2*>(Ar + j * 4) = make_uint2(lo, hi);
    }
    for (int j = par; j < 7; j += 2)                   // zero pad cols 100..127
        *reinterpret_cast<uint2*>(Ar + FDIM + j * 4) = make_uint2(0u, 0u);

    // zero K2 pad region of hidden tile (cols HPtiles*16 .. K2P-1)
    int padStart = HPtiles * 16;
    int padW = K2P - padStart;
    for (int idx = lane; idx < 16 * padW; idx += 32) {
        int r = idx / padW, c = idx - r * padW;
        Hb[r * K2P + padStart + c] = 0;
    }
    lds_fence();

    int l15 = lane & 15;
    int mhi = ((lane >> 4) & 1) << 3;

    // ---- GEMM1: h = relu(in @ w1 + b1) -> bf16 hidden tile in LDS ----
    {
        floatx8 acc[OUT_NT] = {};
        const unsigned short* Arow = A + l15 * NODE_KP;
        for (int k = 0; k < 4; ++k) {
            bf16x16 a = load_frag(Arow, k * 32, lane);
            for (int t = 0; t < HPtiles; ++t) {
                bf16x16 b = load_frag(w1T + (long)(t * 16 + l15) * NODE_KP, k * 32, lane);
                acc[t] = __builtin_amdgcn_wmma_f32_16x16x32_bf16(
                    false, a, false, b, (short)0, acc[t], false, false);
            }
        }
        for (int t = 0; t < HPtiles; ++t) {
            int col = t * 16 + l15;
            float bv = (col < H) ? b1[col] : 0.f;
#pragma unroll
            for (int i = 0; i < 8; ++i) {
                int m = i + mhi;
                float h = acc[t][i] + bv;
                h = h > 0.f ? h : 0.f;
                Hb[m * K2P + col] = f2bf(h);
            }
        }
    }
    lds_fence();

    // ---- GEMM2: y = relu(h @ w2 + b2) -> global ----
    {
        floatx8 acc[OUT_NT] = {};
        const unsigned short* Hrow = Hb + l15 * K2P;
        for (int k = 0; k < K2tiles; ++k) {
            bf16x16 a = load_frag(Hrow, k * 32, lane);
#pragma unroll
            for (int t = 0; t < OUT_NT; ++t) {
                bf16x16 b = load_frag(w2T + (long)(t * 16 + l15) * K2P, k * 32, lane);
                acc[t] = __builtin_amdgcn_wmma_f32_16x16x32_bf16(
                    false, a, false, b, (short)0, acc[t], false, false);
            }
        }
#pragma unroll
        for (int t = 0; t < OUT_NT; ++t) {
            int col = t * 16 + l15;
            if (col < FDIM) {
                float bv = b2[col];
#pragma unroll
                for (int i = 0; i < 8; ++i) {
                    int m = i + mhi;
                    float y = acc[t][i] + bv;
                    y = y > 0.f ? y : 0.f;
                    out[(long)(tile * 16 + m) * FDIM + col] = y;
                }
            }
        }
    }
}

// ------------------------------- launcher ----------------------------------
extern "C" void kernel_launch(void* const* d_in, const int* in_sizes, int n_in,
                              void* d_out, int out_size, void* d_ws, size_t ws_size,
                              hipStream_t stream) {
    const float* x      = (const float*)d_in[0];
    const int*   ei     = (const int*)d_in[1];          // [2][E] (src row, dst row)
    const float* ea     = (const float*)d_in[2];
    const float* lin1_w = (const float*)d_in[3];
    const float* lin1_b = (const float*)d_in[4];
    const float* m1w1   = (const float*)d_in[5];
    const float* m1b1   = (const float*)d_in[6];
    const float* m1w2   = (const float*)d_in[7];
    const float* m1b2   = (const float*)d_in[8];
    const float* lin2_w = (const float*)d_in[9];
    const float* lin2_b = (const float*)d_in[10];
    const float* m2w1   = (const float*)d_in[11];
    const float* m2b1   = (const float*)d_in[12];
    const float* m2w2   = (const float*)d_in[13];
    const float* m2b2   = (const float*)d_in[14];
    const int* src = ei;
    const int* dst = ei + N_EDGES;

    // workspace layout (bytes)
    char* ws = (char*)d_ws;
    unsigned short* lin1T = (unsigned short*)(ws + 0);          // 112*320*2 = 71680
    unsigned short* lin2T = (unsigned short*)(ws + 71680);      // 71680
    unsigned short* w1aT  = (unsigned short*)(ws + 143360);     // 112*128*2 = 28672
    unsigned short* w2aT  = (unsigned short*)(ws + 172032);     // 28672
    unsigned short* w1bT  = (unsigned short*)(ws + 200704);     // 64*128*2  = 16384
    unsigned short* w2bT  = (unsigned short*)(ws + 217088);     // 112*64*2  = 14336
    float* agg = (float*)(ws + 231424);                         // N*F*4 = 20,000,000
    float* f1  = (float*)(ws + 231424 + 20000000);              // N*F*4
    (void)in_sizes; (void)n_in; (void)out_size; (void)ws_size;

    const int THR = 256;
    auto blocks = [](int total, int thr) { return (total + thr - 1) / thr; };

    // 1) convert weights to transposed/padded bf16
    convert_wT_kernel<<<blocks(112 * 320, THR), THR, 0, stream>>>(lin1_w, lin1T, 300, 100, 320, 112);
    convert_wT_kernel<<<blocks(112 * 320, THR), THR, 0, stream>>>(lin2_w, lin2T, 300, 100, 320, 112);
    convert_wT_kernel<<<blocks(112 * 128, THR), THR, 0, stream>>>(m1w1, w1aT, 100, 100, 128, 112);
    convert_wT_kernel<<<blocks(112 * 128, THR), THR, 0, stream>>>(m1w2, w2aT, 100, 100, 128, 112);
    convert_wT_kernel<<<blocks(64 * 128, THR), THR, 0, stream>>>(m2w1, w1bT, 100, 64, 128, 64);
    convert_wT_kernel<<<blocks(112 * 64, THR), THR, 0, stream>>>(m2w2, w2bT, 64, 100, 64, 112);

    const int n4 = N_NODES * FDIM / 4;   // 1,250,000 float4

    // ---- layer 1 ----
    copy4_kernel<<<blocks(n4, THR), THR, 0, stream>>>(x, agg, n4);            // agg = x
    edge_msg_kernel<<<N_EDGES / 16 / 4, 128, 0, stream>>>(x, src, dst, ea, lin1T, lin1_b, agg);
    node_mlp_kernel<<<(N_NODES / 16 + 3) / 4, 128, 0, stream>>>(
        agg, w1aT, m1b1, w2aT, m1b2, f1, /*H=*/100, /*HPtiles=*/7, /*K2P=*/128, /*K2tiles=*/4);

    // ---- layer 2 ----
    copy4_kernel<<<blocks(n4, THR), THR, 0, stream>>>(f1, agg, n4);           // agg = f1
    edge_msg_kernel<<<N_EDGES / 16 / 4, 128, 0, stream>>>(f1, src, dst, ea, lin2T, lin2_b, agg);
    node_mlp_kernel<<<(N_NODES / 16 + 3) / 4, 128, 0, stream>>>(
        agg, w1bT, m2b1, w2bT, m2b2, (float*)d_out, /*H=*/64, /*HPtiles=*/4, /*K2P=*/64, /*K2tiles=*/2);
}
